// MLPRegressor_41815801593928
// MI455X (gfx1250) — matrix operand order repaired and checked by reading, not compiled, not run on hardware
//
#include <hip/hip_runtime.h>

typedef _Float16 v16h __attribute__((ext_vector_type(16)));
typedef _Float16 half8 __attribute__((ext_vector_type(8)));
typedef float    v8f  __attribute__((ext_vector_type(8)));

#define SEQ 256
#define NTHREADS 256

__device__ __forceinline__ void ldcpy(float* dst, const float* src, int n, int tid) {
    for (int i = tid; i < n; i += NTHREADS) dst[i] = src[i];
}

__global__ __launch_bounds__(NTHREADS)
void mlp_fused_kernel(const float* __restrict__ cont_p, const float* __restrict__ cont_c,
                      const int*   __restrict__ cat_p,  const int*   __restrict__ cat_c,
                      const int*   __restrict__ lengths,
                      const float* __restrict__ w_p1, const float* __restrict__ b_p1,
                      const float* __restrict__ w_p2, const float* __restrict__ b_p2,
                      const float* __restrict__ w_c1, const float* __restrict__ b_c1,
                      const float* __restrict__ w_c2, const float* __restrict__ b_c2,
                      const float* __restrict__ eg, const float* __restrict__ ek,
                      const float* __restrict__ epr, const float* __restrict__ ej,
                      const float* __restrict__ er, const float* __restrict__ epl,
                      const float* __restrict__ ea,
                      const float* __restrict__ w_fc1, const float* __restrict__ b_fc1,
                      const float* __restrict__ w_fc2, const float* __restrict__ b_fc2,
                      float* __restrict__ out)
{
    // ---- LDS (~57 KB total) ----
    __shared__ float tab[101 * 32];           // concatenated embedding tables
    __shared__ unsigned int hist[101];        // per-block category histogram
    __shared__ float Wp1[3 * 32], Bp1[32], Wc1[2 * 32], Bc1[32];
    __shared__ float Wp2[32 * 32], Bp2[32], Wc2[32 * 32], Bc2[32];
    __shared__ alignas(16) _Float16 Ap[SEQ * 32];   // masked relu(layer1_p), f16
    __shared__ alignas(16) _Float16 Ac[SEQ * 32];   // masked relu(layer1_c), f16
    __shared__ float pooled[128];
    __shared__ float hmid[64];

    const int tid = threadIdx.x;
    const int b   = blockIdx.x;
    const int len = lengths[b];

    // ---- stage tables + weights into LDS ----
    ldcpy(tab +  0 * 32, eg,   2 * 32, tid);
    ldcpy(tab +  2 * 32, ek,   2 * 32, tid);
    ldcpy(tab +  4 * 32, epr,  2 * 32, tid);
    ldcpy(tab +  6 * 32, ej,  11 * 32, tid);
    ldcpy(tab + 17 * 32, er,  34 * 32, tid);
    ldcpy(tab + 51 * 32, epl, 19 * 32, tid);
    ldcpy(tab + 70 * 32, ea,  31 * 32, tid);
    ldcpy(Wp1, w_p1, 3 * 32, tid);  ldcpy(Bp1, b_p1, 32, tid);
    ldcpy(Wc1, w_c1, 2 * 32, tid);  ldcpy(Bc1, b_c1, 32, tid);
    ldcpy(Wp2, w_p2, 32 * 32, tid); ldcpy(Bp2, b_p2, 32, tid);
    ldcpy(Wc2, w_c2, 32 * 32, tid); ldcpy(Bc2, b_c2, 32, tid);
    if (tid < 101) hist[tid] = 0u;
    if (tid < 128) pooled[tid] = 0.0f;
    __syncthreads();

    // ---- phase 1: per-token layer-1 MLPs (VALU) + category histogram ----
    {
        const int s = tid;
        const long base = (long)b * SEQ + s;
        const float mf = (s < len) ? 1.0f : 0.0f;

        __builtin_prefetch(&cont_p[base * 3], 0, 0);
        __builtin_prefetch(&cat_p[base * 5], 0, 0);

        const float c0 = cont_p[base * 3 + 0];
        const float c1 = cont_p[base * 3 + 1];
        const float c2 = cont_p[base * 3 + 2];
        const float d0 = cont_c[base * 2 + 0];
        const float d1 = cont_c[base * 2 + 1];

        #pragma unroll
        for (int j = 0; j < 32; ++j) {
            float t = Bp1[j];
            t = fmaf(c0, Wp1[j], t);
            t = fmaf(c1, Wp1[32 + j], t);
            t = fmaf(c2, Wp1[64 + j], t);
            t = fmaxf(t, 0.0f);
            Ap[s * 32 + j] = (_Float16)(t * mf);
            float u = Bc1[j];
            u = fmaf(d0, Wc1[j], u);
            u = fmaf(d1, Wc1[32 + j], u);
            u = fmaxf(u, 0.0f);
            Ac[s * 32 + j] = (_Float16)(u * mf);
        }
        if (s < len) {
            atomicAdd(&hist[ 0 + cat_p[base * 5 + 0]], 1u);
            atomicAdd(&hist[ 2 + cat_p[base * 5 + 1]], 1u);
            atomicAdd(&hist[ 4 + cat_p[base * 5 + 2]], 1u);
            atomicAdd(&hist[ 6 + cat_p[base * 5 + 3]], 1u);
            atomicAdd(&hist[17 + cat_p[base * 5 + 4]], 1u);
            atomicAdd(&hist[51 + cat_c[base * 2 + 0]], 1u);
            atomicAdd(&hist[70 + cat_c[base * 2 + 1]], 1u);
        }
    }
    __syncthreads();

    // ---- phase 2: layer-2 GEMMs via v_wmma_f32_16x16x32_f16, pooled in LDS ----
    {
        const int lane  = tid & 31;
        const int wv    = tid >> 5;          // 8 waves, 32 tokens each
        const int m0    = wv * 32;
        const int nlo   = lane & 15;         // row (A) / column (B,D) index
        const int khalf = (lane < 16) ? 0 : 1;

        // B fragments: lane = column n, K split by half-wave, 2 K-values per VGPR
        v16h bp0, bp1, bc0, bc1;
        {
            const int kb = khalf * 16;
            #pragma unroll
            for (int v = 0; v < 8; ++v) {
                bp0[2 * v]     = (_Float16)Wp2[(kb + 2 * v)     * 32 + nlo];
                bp0[2 * v + 1] = (_Float16)Wp2[(kb + 2 * v + 1) * 32 + nlo];
                bp1[2 * v]     = (_Float16)Wp2[(kb + 2 * v)     * 32 + 16 + nlo];
                bp1[2 * v + 1] = (_Float16)Wp2[(kb + 2 * v + 1) * 32 + 16 + nlo];
                bc0[2 * v]     = (_Float16)Wc2[(kb + 2 * v)     * 32 + nlo];
                bc0[2 * v + 1] = (_Float16)Wc2[(kb + 2 * v + 1) * 32 + nlo];
                bc1[2 * v]     = (_Float16)Wc2[(kb + 2 * v)     * 32 + 16 + nlo];
                bc1[2 * v + 1] = (_Float16)Wc2[(kb + 2 * v + 1) * 32 + 16 + nlo];
            }
        }

        v8f accP0 = {}, accP1 = {}, accC0 = {}, accC1 = {};
        const int koff = khalf * 16;  // in halfs

        #pragma unroll
        for (int tm = 0; tm < 2; ++tm) {
            const int m = m0 + tm * 16 + nlo;
            const _Float16* rp = &Ap[m * 32];
            const _Float16* rc = &Ac[m * 32];
            half8 plo = *(const half8*)(rp + koff);        // K 0..7  or 16..23
            half8 phi = *(const half8*)(rp + koff + 8);    // K 8..15 or 24..31
            half8 clo = *(const half8*)(rc + koff);
            half8 chi = *(const half8*)(rc + koff + 8);
            v16h ap = __builtin_shufflevector(plo, phi, 0,1,2,3,4,5,6,7,8,9,10,11,12,13,14,15);
            v16h ac = __builtin_shufflevector(clo, chi, 0,1,2,3,4,5,6,7,8,9,10,11,12,13,14,15);
            accP0 = __builtin_amdgcn_wmma_f32_16x16x32_f16(false, ap, false, bp0, (short)0, accP0, false, false);
            accP1 = __builtin_amdgcn_wmma_f32_16x16x32_f16(false, ap, false, bp1, (short)0, accP1, false, false);
            accC0 = __builtin_amdgcn_wmma_f32_16x16x32_f16(false, ac, false, bc0, (short)0, accC0, false, false);
            accC1 = __builtin_amdgcn_wmma_f32_16x16x32_f16(false, ac, false, bc1, (short)0, accC1, false, false);
        }

        // D layout: VGPR v holds rows v (lanes 0-15) / v+8 (lanes 16-31), N = lane%16 (+16*ntile)
        float sP0 = 0.f, sP1 = 0.f, sC0 = 0.f, sC1 = 0.f;
        #pragma unroll
        for (int v = 0; v < 8; ++v) {
            sP0 += accP0[v]; sP1 += accP1[v]; sC0 += accC0[v]; sC1 += accC1[v];
        }
        atomicAdd(&pooled[64 + nlo],      sP0);
        atomicAdd(&pooled[64 + 16 + nlo], sP1);
        atomicAdd(&pooled[96 + nlo],      sC0);
        atomicAdd(&pooled[96 + 16 + nlo], sC1);
    }
    __syncthreads();

    // ---- phase 3: embedding GEMV from histogram + len*bias for layer-2 ----
    if (tid < 32) {
        const int j = tid;
        float gp = 0.0f, gc = 0.0f;
        for (int v = 0; v < 51; ++v)   gp = fmaf((float)hist[v], tab[v * 32 + j], gp);
        for (int v = 51; v < 101; ++v) gc = fmaf((float)hist[v], tab[v * 32 + j], gc);
        pooled[j]      = gp * 0.2f;   // /5
        pooled[32 + j] = gc * 0.5f;   // /2
        const float fl = (float)len;
        pooled[64 + j] += fl * Bp2[j];
        pooled[96 + j] += fl * Bc2[j];
    }
    __syncthreads();
    if (tid < 128) pooled[tid] *= (1.0f / (float)len);
    __syncthreads();

    // ---- head: fc1 (128->64) relu, fc2 (64->2) relu ----
    if (tid < 64) {
        float acc = b_fc1[tid];
        for (int f = 0; f < 128; ++f) acc = fmaf(pooled[f], w_fc1[f * 64 + tid], acc);
        hmid[tid] = fmaxf(acc, 0.0f);
    }
    __syncthreads();
    if (tid < 2) {
        float acc = b_fc2[tid];
        for (int o = 0; o < 64; ++o) acc = fmaf(hmid[o], w_fc2[o * 2 + tid], acc);
        out[(long)b * 2 + tid] = fmaxf(acc, 0.0f);
    }
}

extern "C" void kernel_launch(void* const* d_in, const int* in_sizes, int n_in,
                              void* d_out, int out_size, void* d_ws, size_t ws_size,
                              hipStream_t stream) {
    const float* cont_p = (const float*)d_in[0];
    const float* cont_c = (const float*)d_in[1];
    const int*   cat_p  = (const int*)d_in[2];
    const int*   cat_c  = (const int*)d_in[3];
    const int*   lens   = (const int*)d_in[4];
    const float* w_p1 = (const float*)d_in[5];  const float* b_p1 = (const float*)d_in[6];
    const float* w_p2 = (const float*)d_in[7];  const float* b_p2 = (const float*)d_in[8];
    const float* w_c1 = (const float*)d_in[9];  const float* b_c1 = (const float*)d_in[10];
    const float* w_c2 = (const float*)d_in[11]; const float* b_c2 = (const float*)d_in[12];
    const float* eg  = (const float*)d_in[13];
    const float* ek  = (const float*)d_in[14];
    const float* epr = (const float*)d_in[15];
    const float* ej  = (const float*)d_in[16];
    const float* er  = (const float*)d_in[17];
    const float* epl = (const float*)d_in[18];
    const float* ea  = (const float*)d_in[19];
    const float* w_fc1 = (const float*)d_in[20]; const float* b_fc1 = (const float*)d_in[21];
    const float* w_fc2 = (const float*)d_in[22]; const float* b_fc2 = (const float*)d_in[23];
    float* out = (float*)d_out;

    const int B = in_sizes[4];   // lengths has B elements
    (void)d_ws; (void)ws_size; (void)n_in; (void)out_size;

    mlp_fused_kernel<<<dim3(B), dim3(NTHREADS), 0, stream>>>(
        cont_p, cont_c, cat_p, cat_c, lens,
        w_p1, b_p1, w_p2, b_p2, w_c1, b_c1, w_c2, b_c2,
        eg, ek, epr, ej, er, epl, ea,
        w_fc1, b_fc1, w_fc2, b_fc2, out);
}
